// Block_14894946582938
// MI455X (gfx1250) — matrix operand order repaired and checked by reading, not compiled, not run on hardware
//
#include <hip/hip_runtime.h>

typedef __bf16 bf16;
typedef __attribute__((ext_vector_type(4)))  __bf16 v4bf;
typedef __attribute__((ext_vector_type(8)))  __bf16 v8bf;
typedef __attribute__((ext_vector_type(16))) __bf16 v16bf;
typedef __attribute__((ext_vector_type(8)))  float  v8f;

#define EMB    1024
#define S_LEN  2048
#define BATCH  2
#define ROWS   (BATCH * S_LEN)   // 4096
#define HIDDEN 4096
#define HEADS  16
#define HDIM   64

// ---------- CDNA5 async Global->LDS copy (ASYNCcnt path, ISA ch.08) ----------
// vdst = 32-bit LDS byte offset (low 32 bits of generic shared pointer),
// vaddr = 64-bit global VA. Copy engine writes LDS directly; no VGPR staging.
static __device__ __forceinline__ void async_copy16(const bf16* gsrc, bf16* lds_dst) {
  asm volatile("global_load_async_to_lds_b128 %0, %1, off"
               :: "v"((unsigned)(unsigned long long)lds_dst), "v"(gsrc)
               : "memory");
}
static __device__ __forceinline__ void wait_async0() {
  asm volatile("s_wait_asynccnt 0" ::: "memory");
}

// ---------- WMMA fragment helpers ----------
// A fragment (16x32 bf16): lane = M row (lane&15), data = K {half*8..+7, 16+half*8..+7}.
// Caller passes p = &rowbase[half*8]; low 8 at +0, high 8 at +16.
static __device__ __forceinline__ v16bf ldA(const bf16* p) {
  v8bf lo = *(const v8bf*)p;
  v8bf hi = *(const v8bf*)(p + 16);
  return __builtin_shufflevector(lo, hi, 0,1,2,3,4,5,6,7,8,9,10,11,12,13,14,15);
}
// B fragment (32x16 bf16): lane = N col (lane&15), data = K {half*8..+7, 16+half*8..+7}.
// Caller passes p = &base[(half*8)*ld + col]; K rows strided by ld.
static __device__ __forceinline__ v16bf ldB(const bf16* p, int ld) {
  v16bf r;
#pragma unroll
  for (int s = 0; s < 8; ++s) { r[s] = p[s * ld]; r[8 + s] = p[(16 + s) * ld]; }
  return r;
}
static __device__ __forceinline__ v8f wmma_bf16(v16bf a, v16bf b, v8f c) {
  return __builtin_amdgcn_wmma_f32_16x16x32_bf16(false, a, false, b, (short)0, c, false, false);
}

// ---------- fp32 -> bf16 cast ----------
__global__ __launch_bounds__(256)
void cast_bf16_kernel(const float* __restrict__ in, bf16* __restrict__ out, int n4) {
  const int i = blockIdx.x * 256 + threadIdx.x;
  if (i < n4) {
    const float4 v = ((const float4*)in)[i];
    v4bf ov; ov[0] = (bf16)v.x; ov[1] = (bf16)v.y; ov[2] = (bf16)v.z; ov[3] = (bf16)v.w;
    ((v4bf*)out)[i] = ov;
  }
}

// ---------- LayerNorm (fp32 in, bf16 out), one row of 1024 per block ----------
__global__ __launch_bounds__(256)
void ln_bf16_kernel(const float* __restrict__ x, const float* __restrict__ w,
                    const float* __restrict__ b, bf16* __restrict__ out) {
  __shared__ float rs[256];
  __shared__ float rq[256];
  const int tid = threadIdx.x;
  const size_t row = blockIdx.x;
  const float4 v = ((const float4*)(x + row * EMB))[tid];
  rs[tid] = v.x + v.y + v.z + v.w;
  rq[tid] = v.x * v.x + v.y * v.y + v.z * v.z + v.w * v.w;
  __syncthreads();
  for (int st = 128; st > 0; st >>= 1) {
    if (tid < st) { rs[tid] += rs[tid + st]; rq[tid] += rq[tid + st]; }
    __syncthreads();
  }
  const float mean = rs[0] * (1.0f / EMB);
  const float var  = rq[0] * (1.0f / EMB) - mean * mean;
  const float rstd = rsqrtf(var + 1e-5f);
  const float4 wv = ((const float4*)w)[tid];
  const float4 bv = ((const float4*)b)[tid];
  v4bf ov;
  ov[0] = (bf16)((v.x - mean) * rstd * wv.x + bv.x);
  ov[1] = (bf16)((v.y - mean) * rstd * wv.y + bv.y);
  ov[2] = (bf16)((v.z - mean) * rstd * wv.z + bv.z);
  ov[3] = (bf16)((v.w - mean) * rstd * wv.w + bv.w);
  *(v4bf*)(out + row * EMB + (size_t)tid * 4) = ov;
}

// ---------- Tiled WMMA GEMM: C[M,N] = A[M,K]*B[K,N] (+bias)(+resid)(relu) ----------
// 128x128 block tile, K-step 32, 256 threads = 8 waves (4x2), 32x64 per wave.
#define AP 40    // As row pitch (halfwords): 32 + 8 pad
#define BP 136   // Bs row pitch (halfwords): 128 + 8 pad

static __device__ __forceinline__ void g_load_tile(
    const bf16* __restrict__ A, const bf16* __restrict__ Bw,
    int K, int N, int row0, int col0, int kt, int nkt,
    bf16* __restrict__ As, bf16* __restrict__ Bs, int tid) {
  const bf16* ag = A + (size_t)row0 * K + (size_t)kt * 32;
  const bf16* bg = Bw + (size_t)kt * 32 * N + col0;
#pragma unroll
  for (int r = 0; r < 2; ++r) {
    const int c = tid + r * 256;
    const int arow = c >> 2, ak = (c & 3) << 3;
    async_copy16(ag + (size_t)arow * K + ak, As + arow * AP + ak);
    const int brow = c >> 4, bn = (c & 15) << 3;
    async_copy16(bg + (size_t)brow * N + bn, Bs + brow * BP + bn);
  }
  if (kt + 2 < nkt) {  // speculative prefetch of K+2 tile -> global_prefetch_b8
    __builtin_prefetch(ag + (size_t)(tid >> 2) * K + ((tid & 3) << 3) + 64, 0, 1);
    __builtin_prefetch(bg + (size_t)(64 + (tid >> 4)) * N + ((tid & 15) << 3), 0, 1);
  }
}

__global__ __launch_bounds__(256, 2)
void gemm_bf16_kernel(const bf16* __restrict__ A, const bf16* __restrict__ Bw,
                      const float* __restrict__ bias, const float* __restrict__ resid,
                      float* __restrict__ outF, bf16* __restrict__ outB,
                      int M, int N, int K, int relu) {
  __shared__ __align__(16) bf16 As[2][128 * AP];
  __shared__ __align__(16) bf16 Bs[2][32 * BP];
  const int tid  = threadIdx.x;
  const int lane = tid & 31, wave = tid >> 5;
  const int wm = wave >> 1, wn = wave & 1;
  const int half = lane >> 4, l16 = lane & 15;
  const int row0 = blockIdx.y << 7, col0 = blockIdx.x << 7;
  (void)M;

  v8f acc[2][4];
#pragma unroll
  for (int i = 0; i < 2; ++i)
#pragma unroll
    for (int j = 0; j < 4; ++j)
#pragma unroll
      for (int e = 0; e < 8; ++e) acc[i][j][e] = 0.0f;

  const int nkt = K >> 5;
  g_load_tile(A, Bw, K, N, row0, col0, 0, nkt, As[0], Bs[0], tid);
  wait_async0();
  __syncthreads();
  for (int kt = 0; kt < nkt; ++kt) {
    const int buf = kt & 1;
    if (kt + 1 < nkt)   // async copy of next tile overlaps the 8 WMMAs below
      g_load_tile(A, Bw, K, N, row0, col0, kt + 1, nkt, As[buf ^ 1], Bs[buf ^ 1], tid);
    const bf16* ab = &As[buf][(wm * 32 + l16) * AP + half * 8];
    const bf16* bb = &Bs[buf][(half * 8) * BP + wn * 64 + l16];
    const v16bf af0 = ldA(ab);
    const v16bf af1 = ldA(ab + 16 * AP);
#pragma unroll
    for (int j = 0; j < 4; ++j) {
      const v16bf bfj = ldB(bb + j * 16, BP);
      acc[0][j] = wmma_bf16(af0, bfj, acc[0][j]);
      acc[1][j] = wmma_bf16(af1, bfj, acc[1][j]);
    }
    wait_async0();
    __syncthreads();
  }

  // Epilogue: C layout -> element e: row = e + half*8, col = l16 (per 16x16 tile)
#pragma unroll
  for (int i = 0; i < 2; ++i) {
#pragma unroll
    for (int e = 0; e < 8; ++e) {
      const int r = row0 + wm * 32 + i * 16 + half * 8 + e;
      const size_t ro = (size_t)r * N;
#pragma unroll
      for (int j = 0; j < 4; ++j) {
        const int c = col0 + wn * 64 + j * 16 + l16;
        float v = acc[i][j][e];
        if (bias)  v += bias[c];
        if (resid) v += resid[ro + c];
        if (relu)  v = fmaxf(v, 0.0f);
        if (outF)  outF[ro + c] = v;
        else       outB[ro + c] = (bf16)v;
      }
    }
  }
}

// ---------- Flash attention: one (b, h, 64-row q-tile) per block, 4 waves ----------
#define DP 72  // 64 + 8 pad row pitch (halfwords)
__global__ __launch_bounds__(128, 2)
void attn_kernel(const bf16* __restrict__ Qg_, const bf16* __restrict__ Kg_,
                 const bf16* __restrict__ Vg_, bf16* __restrict__ Og_) {
  __shared__ __align__(16) bf16 Qs[64 * DP];  // [q][d]
  __shared__ __align__(16) bf16 Kt[64 * DP];  // [d][key]   (transposed on load)
  __shared__ __align__(16) bf16 Vs[64 * DP];  // [key][d]
  __shared__ __align__(16) bf16 Ps[64 * DP];  // [q][key]   (P staging, C->A layout)
  const int tid  = threadIdx.x;
  const int lane = tid & 31, wave = tid >> 5;
  const int half = lane >> 4, l16 = lane & 15;
  const int qt = blockIdx.x;
  const size_t headoff = (size_t)blockIdx.z * S_LEN * EMB + (size_t)blockIdx.y * HDIM;
  const bf16* Qg = Qg_ + headoff;
  const bf16* Kg = Kg_ + headoff;
  const bf16* Vg = Vg_ + headoff;
  bf16*       Og = Og_ + headoff;
  const int q0 = qt * 64;
  const int qw = wave * 16;  // this wave's 16 q rows inside the tile

  // Q tile via async copy engine
#pragma unroll
  for (int r = 0; r < 4; ++r) {
    const int c = tid + r * 128;
    const int qr = c >> 3, dc = (c & 7) << 3;
    async_copy16(Qg + (size_t)(q0 + qr) * EMB + dc, Qs + qr * DP + dc);
  }

  float m_[8], l_[8];
  v8f o[4];
#pragma unroll
  for (int e = 0; e < 8; ++e) { m_[e] = -1e30f; l_[e] = 0.0f; }
#pragma unroll
  for (int j = 0; j < 4; ++j)
#pragma unroll
    for (int e = 0; e < 8; ++e) o[j][e] = 0.0f;

  for (int kt2 = 0; kt2 <= qt; ++kt2) {
    const int k0 = kt2 * 64;
    // V row-major via async; K transposed manually -> Kt[d][key]
#pragma unroll
    for (int r = 0; r < 4; ++r) {
      const int c = tid + r * 128;
      const int kr = c >> 3, dc = (c & 7) << 3;
      async_copy16(Vg + (size_t)(k0 + kr) * EMB + dc, Vs + kr * DP + dc);
      const v8bf kv = *(const v8bf*)(Kg + (size_t)(k0 + kr) * EMB + dc);
#pragma unroll
      for (int j = 0; j < 8; ++j) Kt[(dc + j) * DP + kr] = kv[j];
    }
    wait_async0();
    __syncthreads();

    // S = Q * K^T  (contract over d = 64, two K=32 WMMA steps)
    v8f sc[4];
#pragma unroll
    for (int j = 0; j < 4; ++j)
#pragma unroll
      for (int e = 0; e < 8; ++e) sc[j][e] = 0.0f;
    const bf16* qb = Qs + (qw + l16) * DP + half * 8;
#pragma unroll
    for (int kc = 0; kc < 2; ++kc) {
      const v16bf aq = ldA(qb + kc * 32);
#pragma unroll
      for (int j = 0; j < 4; ++j) {
        const v16bf bk = ldB(Kt + (kc * 32 + half * 8) * DP + j * 16 + l16, DP);
        sc[j] = wmma_bf16(aq, bk, sc[j]);
      }
    }

    // Online softmax. Row r = qw + half*8 + e lives in one 16-lane half; col = j*16 + l16.
    float mx[8];
#pragma unroll
    for (int e = 0; e < 8; ++e) mx[e] = m_[e];
    const bool diag = (kt2 == qt);
#pragma unroll
    for (int j = 0; j < 4; ++j) {
      const int key = k0 + j * 16 + l16;
#pragma unroll
      for (int e = 0; e < 8; ++e) {
        float s = sc[j][e] * 0.125f;  // 1/sqrt(64)
        if (diag && key > (q0 + qw + half * 8 + e)) s = -1e30f;
        sc[j][e] = s;
        mx[e] = fmaxf(mx[e], s);
      }
    }
#pragma unroll
    for (int e = 0; e < 8; ++e) {
      mx[e] = fmaxf(mx[e], __shfl_xor(mx[e], 1, 32));
      mx[e] = fmaxf(mx[e], __shfl_xor(mx[e], 2, 32));
      mx[e] = fmaxf(mx[e], __shfl_xor(mx[e], 4, 32));
      mx[e] = fmaxf(mx[e], __shfl_xor(mx[e], 8, 32));
    }
    float alpha[8], rsum[8];
#pragma unroll
    for (int e = 0; e < 8; ++e) {
      alpha[e] = __expf(m_[e] - mx[e]);
      m_[e] = mx[e];
      rsum[e] = 0.0f;
    }
#pragma unroll
    for (int j = 0; j < 4; ++j)
#pragma unroll
      for (int e = 0; e < 8; ++e) {
        const float p = __expf(sc[j][e] - m_[e]);
        rsum[e] += p;
        Ps[(qw + half * 8 + e) * DP + j * 16 + l16] = (bf16)p;
      }
#pragma unroll
    for (int e = 0; e < 8; ++e) {
      rsum[e] += __shfl_xor(rsum[e], 1, 32);
      rsum[e] += __shfl_xor(rsum[e], 2, 32);
      rsum[e] += __shfl_xor(rsum[e], 4, 32);
      rsum[e] += __shfl_xor(rsum[e], 8, 32);
      l_[e] = l_[e] * alpha[e] + rsum[e];
    }
#pragma unroll
    for (int j = 0; j < 4; ++j)
#pragma unroll
      for (int e = 0; e < 8; ++e) o[j][e] *= alpha[e];
    __syncthreads();  // Ps visible (and keep waves in lockstep)

    // O += P * V  (contract over key = 64, two K=32 WMMA steps)
    const bf16* pb = Ps + (qw + l16) * DP + half * 8;
#pragma unroll
    for (int kc = 0; kc < 2; ++kc) {
      const v16bf ap = ldA(pb + kc * 32);
#pragma unroll
      for (int j = 0; j < 4; ++j) {
        const v16bf bv = ldB(Vs + (kc * 32 + half * 8) * DP + j * 16 + l16, DP);
        o[j] = wmma_bf16(ap, bv, o[j]);
      }
    }
    __syncthreads();  // before Kt/Vs/Ps are overwritten next iteration
  }

#pragma unroll
  for (int e = 0; e < 8; ++e) {
    const int qr = q0 + qw + half * 8 + e;
    const float inv = 1.0f / l_[e];
#pragma unroll
    for (int j = 0; j < 4; ++j)
      Og[(size_t)qr * EMB + j * 16 + l16] = (bf16)(o[j][e] * inv);
  }
}

// ---------- host-side orchestration ----------
extern "C" void kernel_launch(void* const* d_in, const int* in_sizes, int n_in,
                              void* d_out, int out_size, void* d_ws, size_t ws_size,
                              hipStream_t stream) {
  (void)in_sizes; (void)n_in; (void)out_size; (void)ws_size;
  const float* x     = (const float*)d_in[0];
  const float* Wq    = (const float*)d_in[1];
  const float* Wk    = (const float*)d_in[2];
  const float* Wv    = (const float*)d_in[3];
  const float* Wo    = (const float*)d_in[4];
  const float* bo    = (const float*)d_in[5];
  const float* ln1_w = (const float*)d_in[6];
  const float* ln1_b = (const float*)d_in[7];
  const float* ln2_w = (const float*)d_in[8];
  const float* ln2_b = (const float*)d_in[9];
  const float* W1    = (const float*)d_in[10];
  const float* b1    = (const float*)d_in[11];
  const float* W2    = (const float*)d_in[12];
  const float* b2    = (const float*)d_in[13];

  char* ws = (char*)d_ws;
  const size_t MB = 1024 * 1024;
  bf16*  Wqb   = (bf16*)(ws + 0 * MB);    // 2 MB
  bf16*  Wkb   = (bf16*)(ws + 2 * MB);    // 2 MB
  bf16*  Wvb   = (bf16*)(ws + 4 * MB);    // 2 MB
  bf16*  Wob   = (bf16*)(ws + 6 * MB);    // 2 MB
  bf16*  W1b   = (bf16*)(ws + 8 * MB);    // 8 MB
  bf16*  W2b   = (bf16*)(ws + 16 * MB);   // 8 MB
  bf16*  xn1   = (bf16*)(ws + 24 * MB);   // 8 MB
  bf16*  Qb    = (bf16*)(ws + 32 * MB);   // 8 MB
  bf16*  Kb    = (bf16*)(ws + 40 * MB);   // 8 MB
  bf16*  Vb    = (bf16*)(ws + 48 * MB);   // 8 MB
  bf16*  attnb = (bf16*)(ws + 56 * MB);   // 8 MB
  float* hbuf  = (float*)(ws + 64 * MB);  // 16 MB
  bf16*  xn2   = (bf16*)(ws + 80 * MB);   // 8 MB
  bf16*  ff1   = (bf16*)(ws + 88 * MB);   // 32 MB  -> 120 MB total

  // Weight casts to bf16
  cast_bf16_kernel<<<1024, 256, 0, stream>>>(Wq, Wqb, EMB * EMB / 4);
  cast_bf16_kernel<<<1024, 256, 0, stream>>>(Wk, Wkb, EMB * EMB / 4);
  cast_bf16_kernel<<<1024, 256, 0, stream>>>(Wv, Wvb, EMB * EMB / 4);
  cast_bf16_kernel<<<1024, 256, 0, stream>>>(Wo, Wob, EMB * EMB / 4);
  cast_bf16_kernel<<<4096, 256, 0, stream>>>(W1, W1b, EMB * HIDDEN / 4);
  cast_bf16_kernel<<<4096, 256, 0, stream>>>(W2, W2b, HIDDEN * EMB / 4);

  // LN1 -> bf16
  ln_bf16_kernel<<<ROWS, 256, 0, stream>>>(x, ln1_w, ln1_b, xn1);

  // Q/K/V projections (bf16 out)
  const dim3 gE(EMB / 128, ROWS / 128);      // (8, 32)
  gemm_bf16_kernel<<<gE, 256, 0, stream>>>(xn1, Wqb, nullptr, nullptr, nullptr, Qb, ROWS, EMB, EMB, 0);
  gemm_bf16_kernel<<<gE, 256, 0, stream>>>(xn1, Wkb, nullptr, nullptr, nullptr, Kb, ROWS, EMB, EMB, 0);
  gemm_bf16_kernel<<<gE, 256, 0, stream>>>(xn1, Wvb, nullptr, nullptr, nullptr, Vb, ROWS, EMB, EMB, 0);

  // Causal flash attention
  attn_kernel<<<dim3(S_LEN / 64, HEADS, BATCH), 128, 0, stream>>>(Qb, Kb, Vb, attnb);

  // h = x + attn @ Wo + bo   (fp32)
  gemm_bf16_kernel<<<gE, 256, 0, stream>>>(attnb, Wob, bo, x, hbuf, nullptr, ROWS, EMB, EMB, 0);

  // LN2 -> bf16
  ln_bf16_kernel<<<ROWS, 256, 0, stream>>>(hbuf, ln2_w, ln2_b, xn2);

  // ff1 = relu(xn2 @ W1 + b1)  (bf16)
  const dim3 gH(HIDDEN / 128, ROWS / 128);   // (32, 32)
  gemm_bf16_kernel<<<gH, 256, 0, stream>>>(xn2, W1b, b1, nullptr, nullptr, ff1, ROWS, HIDDEN, EMB, 1);

  // out = h + ff1 @ W2 + b2   (fp32 -> d_out)
  gemm_bf16_kernel<<<gE, 256, 0, stream>>>(ff1, W2b, b2, hbuf, (float*)d_out, nullptr, ROWS, EMB, HIDDEN, 0);
}
